// Attention_38113539785305
// MI455X (gfx1250) — matrix operand order repaired and checked by reading, not compiled, not run on hardware
//
#include <hip/hip_runtime.h>

// Problem constants (from reference): B=2, H=16, S=2048, DH=64, fp32 in/out.
#define BATCH 2
#define HEADS 16
#define SEQ   2048
#define DHEAD 64
#define PITCH 17            // LDS pitch (17 dwords) -> conflict-free banks

typedef __attribute__((ext_vector_type(16))) __bf16 v16bf;
typedef __attribute__((ext_vector_type(2)))  __bf16 v2bf;
typedef __attribute__((ext_vector_type(8)))  float  v8f;

union ABPack {
    unsigned u[8];
    v16bf    v;
};

// f32 -> bf16 pair packed into one dword (lo = even K, hi = odd K).
// Prefer the single-instruction v_cvt_pk_bf16_f32 path.
__device__ __forceinline__ unsigned pack_bf16(float lo, float hi) {
#if __has_builtin(__builtin_amdgcn_cvt_pk_bf16_f32)
    v2bf t = __builtin_amdgcn_cvt_pk_bf16_f32(lo, hi);
    return __builtin_bit_cast(unsigned, t);
#else
    v2bf t = { (__bf16)lo, (__bf16)hi };
    return __builtin_bit_cast(unsigned, t);
#endif
}

// A-matrix (bf16 16x32) K index for VGPR j within a lane-half (contiguous: k = 16*half + 2j)
__device__ __forceinline__ int a_kidx(int j, int half) {
    return 2 * j + 16 * half;
}

__global__ void attn_wmma_kernel(const float* __restrict__ Q,
                                 const float* __restrict__ K,
                                 const float* __restrict__ V,
                                 const int*   __restrict__ mask,
                                 float* __restrict__ out,
                                 float* __restrict__ attn)
{
    extern __shared__ float smem[];             // 2 waves * SEQ * PITCH floats
    const int lane = threadIdx.x & 31;
    const int wave = threadIdx.x >> 5;
    float* sc = smem + (size_t)wave * SEQ * PITCH;   // this wave's [k][m] score pad

    // One 16-row Q tile per wave.
    const int qtile = blockIdx.x * 2 + wave;         // 0 .. B*H*S/16 - 1
    const int bh    = qtile >> 7;                    // / (S/16) == /128
    const int q0    = (qtile & 127) << 4;
    const int b     = bh >> 4;                       // / HEADS

    const float* Qb = Q + (size_t)bh * SEQ * DHEAD;
    const float* Kb = K + (size_t)bh * SEQ * DHEAD;
    const float* Vb = V + (size_t)bh * SEQ * DHEAD;
    const int*   mb = mask + (size_t)b * SEQ;

    const int mrow = lane & 15;                      // M (A,C rows) or N (B,C cols)
    const int half = lane >> 4;

    // ---- Load Q tile as WMMA A operands (two K=32 chunks over DH=64), f32->bf16 ----
    ABPack qa[2];
    #pragma unroll
    for (int c = 0; c < 2; ++c) {
        const float* qrow = Qb + (size_t)(q0 + mrow) * DHEAD + c * 32;
        #pragma unroll
        for (int j = 0; j < 8; ++j) {
            const int kk = a_kidx(j, half);
            qa[c].u[j] = pack_bf16(qrow[kk], qrow[kk + 1]);
        }
    }

    const float scale = 0.125f;                      // 1/sqrt(64)

    // ---- Phase 1: scores = QK^T * scale, masked(zero-fill), spill to LDS ----
    for (int k0 = 0; k0 < SEQ; k0 += 16) {
        v8f cacc = {};
        #pragma unroll
        for (int c = 0; c < 2; ++c) {
            // B operand: B[kd][n] = K[k0+n][c*32 + kd]; lane n = mrow, kd = 2j + 16*half
            ABPack bb;
            const float* kr = Kb + (size_t)(k0 + mrow) * DHEAD + c * 32 + 16 * half;
            #pragma unroll
            for (int j = 0; j < 8; ++j)
                bb.u[j] = pack_bf16(kr[2 * j], kr[2 * j + 1]);
            cacc = __builtin_amdgcn_wmma_f32_16x16x32_bf16(
                false, qa[c].v, false, bb.v, (short)0, cacc, false, false);
        }
        // mask column k0+n (mask shape [B,1,1,S]); zero-fill where mask==0
        const float mul = mb[k0 + mrow] ? scale : 0.0f;
        #pragma unroll
        for (int r = 0; r < 8; ++r) {
            const int m = r + 8 * half;              // C layout: VGPR r -> row r (+8 for hi half)
            sc[(size_t)(k0 + mrow) * PITCH + m] = cacc[r] * mul;
        }
    }

    // ---- Phase 2: softmax per row + stream normalized probs to global attn ----
    float* arowbase = attn + (size_t)bh * SEQ * SEQ + (size_t)q0 * SEQ;
    for (int m = 0; m < 16; ++m) {
        float mx = -3.402823466e38f;
        for (int k = lane; k < SEQ; k += 32)
            mx = fmaxf(mx, sc[(size_t)k * PITCH + m]);
        #pragma unroll
        for (int off = 16; off >= 1; off >>= 1)
            mx = fmaxf(mx, __shfl_xor(mx, off, 32));

        float sum = 0.0f;
        for (int k = lane; k < SEQ; k += 32) {
            const float e = expf(sc[(size_t)k * PITCH + m] - mx);
            sc[(size_t)k * PITCH + m] = e;
            sum += e;
        }
        #pragma unroll
        for (int off = 16; off >= 1; off >>= 1)
            sum += __shfl_xor(sum, off, 32);

        const float inv = 1.0f / sum;
        float* arow = arowbase + (size_t)m * SEQ;
        for (int k = lane; k < SEQ; k += 32) {
            const float p = sc[(size_t)k * PITCH + m] * inv;
            sc[(size_t)k * PITCH + m] = p;
            __builtin_nontemporal_store(p, arow + k);   // 537 MB stream: keep out of caches
        }
    }

    // ---- Phase 3: out = attn @ V (16 x 2048 @ 2048 x 64), 4 N-tiles over DH ----
    v8f oacc[4] = {{}, {}, {}, {}};
    for (int k0 = 0; k0 < SEQ; k0 += 32) {
        // A operand: probs[M][k0+kk] from LDS
        ABPack pa;
        #pragma unroll
        for (int j = 0; j < 8; ++j) {
            const int kk = a_kidx(j, half);
            pa.u[j] = pack_bf16(sc[(size_t)(k0 + kk) * PITCH + mrow],
                                sc[(size_t)(k0 + kk + 1) * PITCH + mrow]);
        }
        #pragma unroll
        for (int dt = 0; dt < 4; ++dt) {
            // B operand: B[kd][n] = V[k0+kd][dt*16 + n]; kd = 2j + 16*half, n = mrow
            ABPack vb;
            const float* vr = Vb + (size_t)(k0 + 16 * half) * DHEAD + dt * 16 + mrow;
            #pragma unroll
            for (int j = 0; j < 8; ++j)
                vb.u[j] = pack_bf16(vr[(size_t)(2 * j) * DHEAD],
                                    vr[(size_t)(2 * j + 1) * DHEAD]);
            oacc[dt] = __builtin_amdgcn_wmma_f32_16x16x32_bf16(
                false, pa.v, false, vb.v, (short)0, oacc[dt], false, false);
        }
    }

    // ---- Write out tile [16 x 64] ----
    float* ob = out + (size_t)bh * SEQ * DHEAD + (size_t)q0 * DHEAD;
    #pragma unroll
    for (int dt = 0; dt < 4; ++dt) {
        #pragma unroll
        for (int r = 0; r < 8; ++r) {
            const int m = r + 8 * half;
            ob[(size_t)m * DHEAD + dt * 16 + mrow] = oacc[dt][r];
        }
    }
}

extern "C" void kernel_launch(void* const* d_in, const int* in_sizes, int n_in,
                              void* d_out, int out_size, void* d_ws, size_t ws_size,
                              hipStream_t stream)
{
    const float* Q    = (const float*)d_in[0];
    const float* K    = (const float*)d_in[1];
    const float* V    = (const float*)d_in[2];
    const int*   mask = (const int*)d_in[3];
    // d_in[4] = attention_dropout_prob (0 -> no dropout path), ignored.

    float* out  = (float*)d_out;                                   // [B,H,S,DH]
    float* attn = (float*)d_out + (size_t)BATCH * HEADS * SEQ * DHEAD; // [B,H,S,S]

    const int qtiles = BATCH * HEADS * (SEQ / 16);   // 4096 waves of work
    const dim3 grid(qtiles / 2);                     // 2 waves per block
    const dim3 block(64);
    const size_t shmem = 2ull * SEQ * PITCH * sizeof(float);  // 272 KB / WG

    attn_wmma_kernel<<<grid, block, shmem, stream>>>(Q, K, V, mask, out, attn);
}
